// EgoPredictor_79190607003821
// MI455X (gfx1250) — matrix operand order repaired and checked by reading, not compiled, not run on hardware
//
#include <hip/hip_runtime.h>
#include <hip/hip_bf16.h>
#include <math.h>

// ---------------- problem constants ----------------
constexpr int B_   = 128;
constexpr int N_   = 64;
constexpr int CAP_ = 32;
constexpr int TH_  = 8;
constexpr int TF_  = 12;
constexpr int DIM_ = 2;
constexpr int D_   = 128;
constexpr int DFF_ = 128;
constexpr int H_   = 2;
constexpr int INS_ = 20;
constexpr int DZ_  = 16;
constexpr int PB_  = 2 * B_ * CAP_;       // 8192 packed sequences
constexpr int MTOK = PB_ * TH_;           // 65536 tokens
constexpr int MHALF = MTOK / 2;           // 32768 tokens per half

// closed-form least-squares projection matrices (A = [1, t], t = 0..7)
__device__ __forceinline__ float pfit(int t, int h) {
  return (140.f - 28.f * (float)h + (float)t * (-28.f + 8.f * (float)h)) * (1.f / 336.f);
}
__device__ __forceinline__ float ppred(int f, int h) {
  return (140.f - 28.f * (float)h + (float)(8 + f) * (-28.f + 8.f * (float)h)) * (1.f / 336.f);
}

// ---------------- WMMA types ----------------
typedef __attribute__((ext_vector_type(16))) _Float16 v16h;
typedef __attribute__((ext_vector_type(8)))  _Float16 h8;
typedef __attribute__((ext_vector_type(2)))  __fp16   fp16x2;   // cvt_pkrtz result type
typedef __attribute__((ext_vector_type(8)))  float    v8f;
union F16x16 { v16h v; _Float16 h[16]; fp16x2 d[8]; h8 p[2]; };
union F32x8  { v8f  v; float    f[8];  };

// =====================================================================
// Weight repack: W[K=128,N] fp32 row-major -> Wt[Npad][128] fp16
// (column-major, zero-padded cols) so the WMMA B-fragment per lane is
// 16 contiguous halves -> two b128 loads, no guards in the GEMM.
// =====================================================================
__global__ void wconv_kernel(const float* __restrict__ W, _Float16* __restrict__ Wt,
                             int N, int Npad) {
  int i = blockIdx.x * blockDim.x + threadIdx.x;
  if (i >= Npad * 128) return;
  int n = i >> 7;                    // padded column
  int k = i & 127;
  float v = (n < N) ? W[(size_t)k * N + n] : 0.f;
  Wt[(size_t)n * 128 + k] = (_Float16)v;
}

// =====================================================================
// C[M,N] = act(A[M,128] @ W + bias); A fp32, W pre-packed fp16.
// One wave = one 16x16 tile, 4 waves/block along M, K=128 fully unrolled.
// M must be a multiple of 16 (true for all call sites).
// act: 0=none 1=relu 2=tanh
// =====================================================================
__launch_bounds__(128)
__global__ void gemm_kernel(const float* __restrict__ A, const _Float16* __restrict__ Wt,
                            const float* __restrict__ bias, float* __restrict__ C,
                            int M, int N, int act) {
  constexpr int K = 128;
  const int lane = threadIdx.x & 31;
  const int wave = threadIdx.x >> 5;
  const int tN = blockIdx.x * 16;
  const int tM = (blockIdx.y * 4 + wave) * 16;
  if (tM >= M) return;               // wave-uniform
  const int lh = lane & 15;
  const int hi = (lane >> 4) & 1;
  const int row = tM + lh;           // always < M
  const int col = tN + lh;           // Wt padded -> always loadable
  const float4* __restrict__ Arow = (const float4*)(A + (size_t)row * K);
  const h8*     __restrict__ Bcol = (const h8*)(Wt + (size_t)col * K);

  F32x8 acc;
#pragma unroll
  for (int r = 0; r < 8; ++r) acc.f[r] = 0.f;

#pragma unroll
  for (int k0 = 0; k0 < K; k0 += 32) {
    if (k0 + 32 < K)
      __builtin_prefetch((const void*)(Arow + (k0 >> 2) + 8), 0, 1);  // global_prefetch_b8

    // A fragment: lane holds 16 contiguous K {ka..ka+15}
    const int qa = (k0 >> 2) + hi * 4;         // float4 index of ka = k0 + hi*16
    float4 a0 = Arow[qa];
    float4 a1 = Arow[qa + 1];
    float4 a2 = Arow[qa + 2];                  // ka + 8
    float4 a3 = Arow[qa + 3];
    F16x16 a;
    a.d[0] = __builtin_amdgcn_cvt_pkrtz(a0.x, a0.y);
    a.d[1] = __builtin_amdgcn_cvt_pkrtz(a0.z, a0.w);
    a.d[2] = __builtin_amdgcn_cvt_pkrtz(a1.x, a1.y);
    a.d[3] = __builtin_amdgcn_cvt_pkrtz(a1.z, a1.w);
    a.d[4] = __builtin_amdgcn_cvt_pkrtz(a2.x, a2.y);
    a.d[5] = __builtin_amdgcn_cvt_pkrtz(a2.z, a2.w);
    a.d[6] = __builtin_amdgcn_cvt_pkrtz(a3.x, a3.y);
    a.d[7] = __builtin_amdgcn_cvt_pkrtz(a3.z, a3.w);

    // B fragment: lane = column, 16 contiguous halves at kb = k0 + hi*16
    const int qb = (k0 >> 3) + hi * 2;         // h8 index
    F16x16 b;
    b.p[0] = Bcol[qb];
    b.p[1] = Bcol[qb + 1];

    acc.v = __builtin_amdgcn_wmma_f32_16x16x32_f16(false, a.v, false, b.v,
                                                   (short)0, acc.v, false, false);
  }

  // epilogue: VGPR r holds row (tM + hi*8 + r), col = tN + lh
  const bool colOK = col < N;
  const int rbase = tM + hi * 8;
  const float bv = (colOK && bias) ? bias[col] : 0.f;
#pragma unroll
  for (int r = 0; r < 8; ++r) {
    if (colOK) {
      float v = acc.f[r] + bv;
      if (act == 1)      v = fmaxf(v, 0.f);
      else if (act == 2) v = tanhf(v);
      C[(size_t)(rbase + r) * N + col] = v;
    }
  }
}

// =====================================================================
// top-CAP nearest neighbors per ego (stable smallest-dist, idx sorted asc)
// =====================================================================
__global__ void topk_kernel(const float* __restrict__ x_ego, const float* __restrict__ x_nei,
                            int* __restrict__ idx) {
  int bi = blockIdx.x;
  __shared__ float dist[N_];
  __shared__ int chosen[CAP_];
  int t = threadIdx.x;
  if (t < N_) {
    float ex = x_ego[((size_t)bi * TH_ + 7) * 2 + 0];
    float ey = x_ego[((size_t)bi * TH_ + 7) * 2 + 1];
    float nx = x_nei[(((size_t)bi * N_ + t) * TH_ + 7) * 2 + 0];
    float ny = x_nei[(((size_t)bi * N_ + t) * TH_ + 7) * 2 + 1];
    float dx = ex - nx, dy = ey - ny;
    dist[t] = dx * dx + dy * dy;
  }
  __syncthreads();
  if (t == 0) {
    bool used[N_];
    for (int j = 0; j < N_; ++j) used[j] = false;
    for (int s = 0; s < CAP_; ++s) {
      float best = 3.4e38f; int bj = 0;
      for (int j = 0; j < N_; ++j)
        if (!used[j] && dist[j] < best) { best = dist[j]; bj = j; }
      used[bj] = true; chosen[s] = bj;
    }
    for (int a = 1; a < CAP_; ++a) {
      int key = chosen[a]; int q = a - 1;
      while (q >= 0 && chosen[q] > key) { chosen[q + 1] = chosen[q]; --q; }
      chosen[q + 1] = key;
    }
    for (int s = 0; s < CAP_; ++s) idx[bi * CAP_ + s] = chosen[s];
  }
}

// =====================================================================
// xp = packed - ref; x_diff = xp - P_FIT@xp; y_lin = P_PRED@xp
// =====================================================================
__global__ void prep_kernel(const float* __restrict__ x_ego, const float* __restrict__ x_nei,
                            const int* __restrict__ idx, float* __restrict__ ref,
                            float* __restrict__ xdiff, float* __restrict__ ylin) {
  int pb = blockIdx.x;
  __shared__ float xp[TH_][2];
  int t = threadIdx.x;               // 32
  if (t < TH_ * 2) {
    int hh = t >> 1, d = t & 1;
    float v;
    if (pb < B_ * CAP_) {
      int bi = pb / CAP_;
      v = x_ego[((size_t)bi * TH_ + hh) * 2 + d];
    } else {
      int e = pb - B_ * CAP_;
      int bi = e / CAP_, c = e % CAP_;
      int n = idx[bi * CAP_ + c];
      v = x_nei[(((size_t)bi * N_ + n) * TH_ + hh) * 2 + d];
    }
    xp[hh][d] = v;
  }
  __syncthreads();
  float r0 = xp[TH_ - 1][0], r1 = xp[TH_ - 1][1];
  __syncthreads();
  if (t < TH_ * 2) {
    int hh = t >> 1, d = t & 1;
    xp[hh][d] -= (d ? r1 : r0);
  }
  if (t < 2) ref[(size_t)pb * 2 + t] = (t ? r1 : r0);
  __syncthreads();
  if (t < TH_ * 2) {
    int tt = t >> 1, d = t & 1;
    float lin = 0.f;
    for (int h = 0; h < TH_; ++h) lin += pfit(tt, h) * xp[h][d];
    xdiff[((size_t)pb * TH_ + tt) * 2 + d] = xp[tt][d] - lin;
  }
  if (t < TF_ * 2) {
    int f = t >> 1, d = t & 1;
    float v = 0.f;
    for (int h = 0; h < TH_; ++h) v += ppred(f, h) * xp[h][d];
    ylin[((size_t)pb * TF_ + f) * 2 + d] = v;
  }
}

// =====================================================================
// src = concat(relu(x_diff@W_fd+b), tanh(z@W_z+b))  -> [MTOK,128]
// =====================================================================
__global__ void feat_kernel(const float* __restrict__ xd, const float* __restrict__ z,
                            const float* __restrict__ W_fd, const float* __restrict__ b_fd,
                            const float* __restrict__ W_z, const float* __restrict__ b_z,
                            float* __restrict__ SRC) {
  int tok = blockIdx.x;
  int c = threadIdx.x;               // 128
  float v;
  if (c < 64) {
    v = xd[(size_t)tok * 2 + 0] * W_fd[c] + xd[(size_t)tok * 2 + 1] * W_fd[64 + c] + b_fd[c];
    v = fmaxf(v, 0.f);
  } else {
    int cc = c - 64;
    float a = b_z[cc];
    const float* zr = z + (size_t)tok * DZ_;
#pragma unroll
    for (int j = 0; j < DZ_; ++j) a += zr[j] * W_z[j * 64 + cc];
    v = tanhf(a);
  }
  SRC[(size_t)tok * D_ + c] = v;
}

__device__ __forceinline__ float pos_enc(int t, int c) {
  float expo = (float)(2 * (c >> 1)) / (float)D_;
  float ang = (float)t * powf(10000.f, -expo);
  return (c & 1) ? cosf(ang) : sinf(ang);
}

__global__ void embed_src_post(float* __restrict__ X) {
  size_t i = (size_t)blockIdx.x * blockDim.x + threadIdx.x;
  if (i >= (size_t)MTOK * D_) return;
  int c = (int)(i & (D_ - 1));
  int t = (int)((i >> 7) & (TH_ - 1));
  X[i] = X[i] * 11.31370850f + pos_enc(t, c);
}

__global__ void embed_tgt_kernel(const float* __restrict__ xd, const float* __restrict__ W_di,
                                 const float* __restrict__ b_di, float* __restrict__ Y) {
  size_t i = (size_t)blockIdx.x * blockDim.x + threadIdx.x;
  if (i >= (size_t)MTOK * D_) return;
  int c = (int)(i & (D_ - 1));
  size_t row = i >> 7;
  int t = (int)(row & (TH_ - 1));
  float v = xd[row * 2 + 0] * W_di[c] + xd[row * 2 + 1] * W_di[D_ + c] + b_di[c];
  Y[i] = v * 11.31370850f + pos_enc(t, c);
}

// =====================================================================
// per-(sequence,head) attention: softmax(QK^T/8)V, T=8, head_dim=64
// =====================================================================
__global__ void attn_kernel(const float* __restrict__ Q, const float* __restrict__ K,
                            const float* __restrict__ V, float* __restrict__ O) {
  int unit = blockIdx.x;
  int s = unit >> 1;
  int h = unit & 1;
  const int HD = D_ / H_;
  size_t base = (size_t)s * TH_ * D_ + (size_t)h * HD;
  __shared__ float sK[TH_][64];
  __shared__ float sV[TH_][64];
  __shared__ float att[TH_][TH_];
  int t = threadIdx.x;               // 64
  for (int i = t; i < TH_ * HD; i += 64) {
    int r = i >> 6, c = i & 63;
    sK[r][c] = K[base + (size_t)r * D_ + c];
    sV[r][c] = V[base + (size_t)r * D_ + c];
  }
  __syncthreads();
  {
    int qi = t >> 3, ki = t & 7;
    float a = 0.f;
    for (int c = 0; c < HD; ++c) a += Q[base + (size_t)qi * D_ + c] * sK[ki][c];
    att[qi][ki] = a * 0.125f;
  }
  __syncthreads();
  if (t < TH_) {
    float m = att[t][0];
    for (int k = 1; k < TH_; ++k) m = fmaxf(m, att[t][k]);
    float ssum = 0.f;
    for (int k = 0; k < TH_; ++k) { float e = expf(att[t][k] - m); att[t][k] = e; ssum += e; }
    float inv = 1.f / ssum;
    for (int k = 0; k < TH_; ++k) att[t][k] *= inv;
  }
  __syncthreads();
  for (int i = t; i < TH_ * HD; i += 64) {
    int qi = i >> 6, c = i & 63;
    float o = 0.f;
    for (int ki = 0; ki < TH_; ++ki) o += att[qi][ki] * sV[ki][c];
    O[base + (size_t)qi * D_ + c] = o;
  }
}

// =====================================================================
// x = LN(x + delta)  — one wave per token
// =====================================================================
__global__ void add_ln_kernel(float* __restrict__ X, const float* __restrict__ Dl,
                              const float* __restrict__ g, const float* __restrict__ bb,
                              int M) {
  int row = blockIdx.x * 4 + (threadIdx.x >> 5);
  if (row >= M) return;
  int lane = threadIdx.x & 31;
  float v[4]; float s = 0.f;
#pragma unroll
  for (int j = 0; j < 4; ++j) {
    int c = lane * 4 + j;
    v[j] = X[(size_t)row * D_ + c] + Dl[(size_t)row * D_ + c];
    s += v[j];
  }
  for (int m = 16; m; m >>= 1) s += __shfl_xor(s, m, 32);
  float mean = s * (1.f / 128.f);
  float var = 0.f;
#pragma unroll
  for (int j = 0; j < 4; ++j) { float d = v[j] - mean; var += d * d; }
  for (int m = 16; m; m >>= 1) var += __shfl_xor(var, m, 32);
  var *= (1.f / 128.f);
  float inv = rsqrtf(var + 1e-6f);
#pragma unroll
  for (int j = 0; j < 4; ++j) {
    int c = lane * 4 + j;
    X[(size_t)row * D_ + c] = g[c] * (v[j] - mean) * inv + bb[c];
  }
}

// =====================================================================
// out[bi,n,k,f,d] = y_base broadcast over k
// =====================================================================
__global__ void fill_base_kernel(const float* __restrict__ x_nei, float* __restrict__ out) {
  int bn = blockIdx.x;
  int bi = bn >> 6, n = bn & 63;
  int t = threadIdx.x;               // 512
  if (t >= INS_ * TF_ * DIM_) return;
  int k = t / (TF_ * DIM_);
  int rem = t % (TF_ * DIM_);
  int f = rem >> 1, d = rem & 1;
  float acc = 0.f;
  const float* xr = x_nei + (((size_t)bi * N_ + n) * TH_) * 2 + d;
#pragma unroll
  for (int h = 0; h < TH_; ++h) acc += ppred(f, h) * xr[h * 2];
  out[(((size_t)bn * INS_ + k) * TF_ + f) * DIM_ + d] = acc;
}

// =====================================================================
// y[e,k,f,d] = sum_h I*R*g + b_dec + y_lin + ref; scatter to out
// =====================================================================
__global__ void head_scatter_kernel(const float* __restrict__ I, const float* __restrict__ R,
                                    const float* __restrict__ g, const float* __restrict__ b_dec,
                                    const float* __restrict__ ylin, const float* __restrict__ ref,
                                    const int* __restrict__ idx, float* __restrict__ out) {
  int e = blockIdx.x;
  int bi = e / CAP_, c = e % CAP_;
  __shared__ float sI[TH_][INS_];
  __shared__ float sR[TH_][TF_];
  __shared__ float sg[TH_][2];
  int t = threadIdx.x;               // 512
  for (int i = t; i < TH_ * INS_; i += 512) sI[i / INS_][i % INS_] = I[(size_t)e * TH_ * INS_ + i];
  for (int i = t; i < TH_ * TF_;  i += 512) sR[i / TF_][i % TF_]   = R[(size_t)e * TH_ * TF_ + i];
  if (t < TH_ * 2) sg[t >> 1][t & 1] = g[(size_t)e * TH_ * 2 + t];
  __syncthreads();
  if (t >= INS_ * TF_ * DIM_) return;
  int k = t / (TF_ * DIM_), rem = t % (TF_ * DIM_);
  int f = rem >> 1, d = rem & 1;
  float acc = 0.f;
#pragma unroll
  for (int h = 0; h < TH_; ++h) acc += sI[h][k] * sR[h][f] * sg[h][d];
  size_t pe = (size_t)(B_ * CAP_ + e);
  acc += b_dec[d] + ylin[pe * TF_ * 2 + (size_t)f * 2 + d] + ref[pe * 2 + d];
  int n = idx[bi * CAP_ + c];
  out[((((size_t)bi * N_ + n) * INS_ + k) * TF_ + f) * DIM_ + d] = acc;
}

// =====================================================================
// host orchestration
// =====================================================================
extern "C" void kernel_launch(void* const* d_in, const int* in_sizes, int n_in,
                              void* d_out, int out_size, void* d_ws, size_t ws_size,
                              hipStream_t stream) {
  (void)in_sizes; (void)n_in; (void)out_size; (void)ws_size;
  const float* x_ego = (const float*)d_in[0];
  const float* x_nei = (const float*)d_in[1];
  const float* z     = (const float*)d_in[2];

  int p = 3;
  auto nxt = [&]() { return (const float*)d_in[p++]; };
  struct MHAW { const float *Wq, *bq, *Wk, *bk, *Wv, *bv, *Wo, *bo; };
  auto nxt_mha = [&]() {
    MHAW m; m.Wq = nxt(); m.bq = nxt(); m.Wk = nxt(); m.bk = nxt();
    m.Wv = nxt(); m.bv = nxt(); m.Wo = nxt(); m.bo = nxt(); return m;
  };

  // params in setup_inputs() insertion order
  const float* W_fd = nxt(); const float* b_fd = nxt();
  const float* W_z  = nxt(); const float* b_z  = nxt();
  const float* W_ei = nxt(); const float* b_ei = nxt();
  const float* W_di = nxt(); const float* b_di = nxt();
  struct EncW { MHAW attn; const float *ln1g,*ln1b,*ln2g,*ln2b,*Wf1,*bf1,*Wf2,*bf2; } enc[2];
  for (int l = 0; l < 2; ++l) {
    enc[l].attn = nxt_mha();
    enc[l].ln1g = nxt(); enc[l].ln1b = nxt(); enc[l].ln2g = nxt(); enc[l].ln2b = nxt();
    enc[l].Wf1 = nxt(); enc[l].bf1 = nxt(); enc[l].Wf2 = nxt(); enc[l].bf2 = nxt();
  }
  struct DecW { MHAW self, cross;
                const float *ln1g,*ln1b,*ln2g,*ln2b,*ln3g,*ln3b,*Wf1,*bf1,*Wf2,*bf2; } dec[2];
  for (int l = 0; l < 2; ++l) {
    dec[l].self = nxt_mha(); dec[l].cross = nxt_mha();
    dec[l].ln1g = nxt(); dec[l].ln1b = nxt(); dec[l].ln2g = nxt(); dec[l].ln2b = nxt();
    dec[l].ln3g = nxt(); dec[l].ln3b = nxt();
    dec[l].Wf1 = nxt(); dec[l].bf1 = nxt(); dec[l].Wf2 = nxt(); dec[l].bf2 = nxt();
  }
  const float* Wk1a = nxt(); const float* bk1a = nxt();
  const float* Wk1b = nxt(); const float* bk1b = nxt();
  const float* Wk2a = nxt(); const float* bk2a = nxt();
  const float* Wk2b = nxt(); const float* bk2b = nxt();
  const float* W_dec = nxt(); const float* b_dec = nxt();

  // ---- workspace carve ----
  char* w = (char*)d_ws;
  auto carve = [&](size_t nfloats) { float* q = (float*)w; w += nfloats * sizeof(float); return q; };
  int*   idx   = (int*)carve(B_ * CAP_);
  float* ref   = carve((size_t)PB_ * 2);
  float* xdiff = carve((size_t)PB_ * TH_ * 2);
  float* ylin  = carve((size_t)PB_ * TF_ * 2);
  float* X  = carve((size_t)MTOK * D_);
  float* Y  = carve((size_t)MTOK * D_);
  float* Qb = carve((size_t)MTOK * D_);   // also SRC features
  float* Kb = carve((size_t)MTOK * D_);
  float* Vb = carve((size_t)MTOK * D_);
  float* T1 = carve((size_t)MTOK * D_);
  float* T2 = carve((size_t)MTOK * D_);
  float* Ibuf = carve((size_t)MHALF * INS_);
  float* Rbuf = carve((size_t)MHALF * TF_);
  float* gbuf = carve((size_t)MHALF * DIM_);

  // fp16 packed-weight arena (carved from the same workspace)
  _Float16* wh = (_Float16*)w;
  auto conv = [&](const float* Wsrc, int N) {
    int Npad = (N + 15) & ~15;
    _Float16* dst = wh;
    wh += (size_t)Npad * 128;
    int total = Npad * 128;
    wconv_kernel<<<(total + 255) / 256, 256, 0, stream>>>(Wsrc, dst, N, Npad);
    return (const _Float16*)dst;
  };

  struct MHAH { const _Float16 *Wq, *Wk, *Wv, *Wo; const float *bq, *bk, *bv, *bo; };
  auto conv_mha = [&](const MHAW& m) {
    MHAH h;
    h.Wq = conv(m.Wq, D_); h.Wk = conv(m.Wk, D_);
    h.Wv = conv(m.Wv, D_); h.Wo = conv(m.Wo, D_);
    h.bq = m.bq; h.bk = m.bk; h.bv = m.bv; h.bo = m.bo;
    return h;
  };

  const _Float16* hW_ei = conv(W_ei, D_);
  MHAH henc_attn[2]; const _Float16* henc_f1[2]; const _Float16* henc_f2[2];
  for (int l = 0; l < 2; ++l) {
    henc_attn[l] = conv_mha(enc[l].attn);
    henc_f1[l] = conv(enc[l].Wf1, DFF_);
    henc_f2[l] = conv(enc[l].Wf2, D_);
  }
  MHAH hdec_self[2], hdec_cross[2]; const _Float16* hdec_f1[2]; const _Float16* hdec_f2[2];
  for (int l = 0; l < 2; ++l) {
    hdec_self[l]  = conv_mha(dec[l].self);
    hdec_cross[l] = conv_mha(dec[l].cross);
    hdec_f1[l] = conv(dec[l].Wf1, DFF_);
    hdec_f2[l] = conv(dec[l].Wf2, D_);
  }
  const _Float16* hWk1a = conv(Wk1a, D_);
  const _Float16* hWk1b = conv(Wk1b, INS_);
  const _Float16* hWk2a = conv(Wk2a, D_);
  const _Float16* hWk2b = conv(Wk2b, TF_);
  const _Float16* hW_dec = conv(W_dec, DIM_);

  auto gemm = [&](const float* A, const _Float16* Wt, const float* bias, float* Cc,
                  int M, int N, int act) {
    int Npad = (N + 15) & ~15;
    dim3 grid(Npad / 16, (M + 63) / 64);
    gemm_kernel<<<grid, 128, 0, stream>>>(A, Wt, bias, Cc, M, N, act);
  };
  auto mha = [&](const float* Xq, const float* Xkv, const MHAH& m, float* outD) {
    gemm(Xq,  m.Wq, m.bq, Qb, MTOK, D_, 0);
    gemm(Xkv, m.Wk, m.bk, Kb, MTOK, D_, 0);
    gemm(Xkv, m.Wv, m.bv, Vb, MTOK, D_, 0);
    attn_kernel<<<PB_ * H_, 64, 0, stream>>>(Qb, Kb, Vb, T1);
    gemm(T1, m.Wo, m.bo, outD, MTOK, D_, 0);
  };
  auto add_ln = [&](float* Xs, const float* Dl, const float* g, const float* bb) {
    add_ln_kernel<<<(MTOK + 3) / 4, 128, 0, stream>>>(Xs, Dl, g, bb, MTOK);
  };

  // ---- stage 0/1: neighbor select, packing, features ----
  topk_kernel<<<B_, 64, 0, stream>>>(x_ego, x_nei, idx);
  prep_kernel<<<PB_, 32, 0, stream>>>(x_ego, x_nei, idx, ref, xdiff, ylin);
  feat_kernel<<<MTOK, 128, 0, stream>>>(xdiff, z, W_fd, b_fd, W_z, b_z, Qb);

  // ---- encoder ----
  gemm(Qb, hW_ei, b_ei, X, MTOK, D_, 0);
  embed_src_post<<<((size_t)MTOK * D_ + 255) / 256, 256, 0, stream>>>(X);
  for (int l = 0; l < 2; ++l) {
    mha(X, X, henc_attn[l], T2);
    add_ln(X, T2, enc[l].ln1g, enc[l].ln1b);
    gemm(X,  henc_f1[l], enc[l].bf1, T1, MTOK, DFF_, 1);
    gemm(T1, henc_f2[l], enc[l].bf2, T2, MTOK, D_, 0);
    add_ln(X, T2, enc[l].ln2g, enc[l].ln2b);
  }

  // ---- decoder ----
  embed_tgt_kernel<<<((size_t)MTOK * D_ + 255) / 256, 256, 0, stream>>>(xdiff, W_di, b_di, Y);
  for (int l = 0; l < 2; ++l) {
    mha(Y, Y, hdec_self[l], T2);
    add_ln(Y, T2, dec[l].ln1g, dec[l].ln1b);
    mha(Y, X, hdec_cross[l], T2);
    add_ln(Y, T2, dec[l].ln2g, dec[l].ln2b);
    gemm(Y,  hdec_f1[l], dec[l].bf1, T1, MTOK, DFF_, 1);
    gemm(T1, hdec_f2[l], dec[l].bf2, T2, MTOK, D_, 0);
    add_ln(Y, T2, dec[l].ln3g, dec[l].ln3b);
  }

  // ---- heads ----
  const float* Ynei = Y + (size_t)MHALF * D_;
  gemm(Y,    hWk1a, bk1a, T2,   MHALF, D_,   1);
  gemm(T2,   hWk1b, bk1b, Ibuf, MHALF, INS_, 2);
  gemm(Ynei, hWk2a, bk2a, T2,   MHALF, D_,   1);
  gemm(T2,   hWk2b, bk2b, Rbuf, MHALF, TF_,  2);
  gemm(Ynei, hW_dec, nullptr, gbuf, MHALF, DIM_, 0);

  // ---- output ----
  fill_base_kernel<<<B_ * N_, 512, 0, stream>>>(x_nei, (float*)d_out);
  head_scatter_kernel<<<B_ * CAP_, 512, 0, stream>>>(Ibuf, Rbuf, gbuf, b_dec,
                                                     ylin, ref, idx, (float*)d_out);
}
